// CenterHeadInf_59304908423224
// MI455X (gfx1250) — matrix operand order repaired and checked by reading, not compiled, not run on hardware
//
#include <hip/hip_runtime.h>
#include <math.h>

typedef __attribute__((ext_vector_type(16))) _Float16 v16h;
typedef __attribute__((ext_vector_type(8)))  _Float16 v8h;
typedef __attribute__((ext_vector_type(8)))  float    v8f;
typedef __attribute__((ext_vector_type(4)))  int      v4i;

#define HH   376
#define WW   376
#define HWSZ (HH*WW)        // 141376
#define CIN0 256
#define CSH  64
#define NCLS 3
#define TOPK 500
#define CAPB 4096

// CDNA5 async global->LDS staging (ASYNCcnt), if the toolchain exposes it.
#if defined(__has_builtin)
#  if __has_builtin(__builtin_amdgcn_global_load_async_to_lds_b128) && \
      __has_builtin(__builtin_amdgcn_s_wait_asynccnt)
#    define USE_ASYNC_LDS 1
#  endif
#endif
#ifndef USE_ASYNC_LDS
#  define USE_ASYNC_LDS 0
#endif

// ---------------- workspace layout (bytes) ----------------
constexpr size_t SZ_XT   = (size_t)HWSZ * CIN0 * 2;      // x transposed f16 [p][ci]
constexpr size_t SZ_SH   = (size_t)HWSZ * CSH  * 2;      // shared_conv out f16 [p][c]
constexpr size_t SZ_BR   = (size_t)HWSZ * CSH  * 2;      // branch mid out  f16 [p][c]
constexpr size_t SZ_OUTS = (size_t)6 * 3 * HWSZ * 4;     // final conv outs fp32 [6][3][HW]
constexpr size_t SZ_WSH  = (size_t)CSH * CIN0 * 9 * 2;   // packed shared weights
constexpr size_t SZ_W1   = (size_t)6 * CSH * CSH * 9 * 2;
constexpr size_t SZ_W2   = (size_t)6 * 16  * CSH * 9 * 2; // M padded 3->16
constexpr size_t SZ_HIST = (size_t)65536 * 4;
constexpr size_t SZ_META = 256;
constexpr size_t SZ_CAND = (size_t)CAPB * 4;

constexpr size_t OFF_XT   = 0;
constexpr size_t OFF_SH   = OFF_XT   + SZ_XT;
constexpr size_t OFF_BR   = OFF_SH   + SZ_SH;
constexpr size_t OFF_OUTS = OFF_BR   + SZ_BR;
constexpr size_t OFF_WSH  = OFF_OUTS + SZ_OUTS;
constexpr size_t OFF_W1   = OFF_WSH  + SZ_WSH;
constexpr size_t OFF_W2   = OFF_W1   + SZ_W1;
constexpr size_t OFF_HIST = OFF_W2   + SZ_W2;
constexpr size_t OFF_META = OFF_HIST + SZ_HIST;
constexpr size_t OFF_CA   = OFF_META + SZ_META;
constexpr size_t OFF_CB   = OFF_CA   + SZ_CAND;

// ---------------- utility: zero u32 region ----------------
__global__ void zero_u32(unsigned* __restrict__ p, int n) {
  int i = blockIdx.x * blockDim.x + threadIdx.x;
  if (i < n) p[i] = 0u;
}

// ---------------- transpose + fp32->f16 convert: x[ci][p] -> xt[p][ci] ----------------
__global__ __launch_bounds__(256)
void transpose_cvt(const float* __restrict__ x, _Float16* __restrict__ xt) {
  __shared__ float tile[32][33];
  const int p0 = blockIdx.x * 32, c0 = blockIdx.y * 32;
  for (int j = 0; j < 4; ++j) {
    int t  = threadIdx.x + 256 * j;
    int pc = t & 31, cr = t >> 5;
    int p  = p0 + pc;
    tile[cr][pc] = (p < HWSZ) ? x[(size_t)(c0 + cr) * HWSZ + p] : 0.f;
  }
  __syncthreads();
  for (int j = 0; j < 4; ++j) {
    int t  = threadIdx.x + 256 * j;
    int cl = t & 31, pr = t >> 5;
    int p  = p0 + pr;
    if (p < HWSZ) xt[(size_t)p * CIN0 + c0 + cl] = (_Float16)tile[cl][pr];
  }
}

// ---------------- weight repack into WMMA A-fragment layout ----------------
// A 16x32 f16 fragment (CDNA5 ISA): lane L holds row M=L&15;
// element e of v16h holds k = e + 16*(L>>4) (this code's contiguous order, not the ISA table order).
// dst layout: [branch][kb][mt][lane][e] contiguous -> per-lane v16h load is 32B coalesced.
// k ordering: k = (dy*3+dx)*CIN + ci  (so a 32-wide k-block has a single (dy,dx)).
__global__ void pack_weights(const float* __restrict__ src, _Float16* __restrict__ dst,
                             int CIN, int MT, int OReal, int perBranch, int nTotal) {
  int i = blockIdx.x * blockDim.x + threadIdx.x;
  if (i >= nTotal) return;
  int bb = i / perBranch, rr = i % perBranch;
  int e = rr & 15, lane = (rr >> 4) & 31, t = rr >> 9;
  int mt = t % MT, kb = t / MT;
  int m  = mt * 16 + (lane & 15);
  int k  = kb * 32 + (e + 16 * (lane >> 4));
  int r  = k / CIN, ci = k % CIN;
  float val = 0.f;
  if (m < OReal) val = src[(((size_t)bb * OReal + m) * CIN + ci) * 9 + r];
  dst[i] = (_Float16)val;
}

// ---------------- implicit-GEMM 3x3 conv via WMMA, 2D register blocking ----------------
// Each wave computes MW x 2 tiles of 16x16: A fragments reused across 2 N-tiles,
// B fragments reused across MW M-tiles.
// A panel (COUT x 32 k-slice) staged per-workgroup into LDS with async DMA
// (global_load_async_to_lds_b128 + s_wait_asynccnt), double buffered.
// in_t: f16 [HW][CIN] (transposed). B fragment: lane holds column N=lane&15,
// k = e + 16*(lane>>4) -> contiguous 16 halves in a row of in_t (v16h load).
// MODE 0: BN + ReLU, f16 transposed out [HW][COUT]
// MODE 1: bias (+optional sigmoid), fp32 out [3][HW] (rows >= NCLS discarded)
template<int CIN, int MT, int MW, int MODE>
__global__ __launch_bounds__(256)
void conv_wmma(const _Float16* __restrict__ in_t, const _Float16* __restrict__ wpack,
               const float* __restrict__ bnp, _Float16* __restrict__ out_t,
               float* __restrict__ out_f, int do_sigmoid) {
  constexpr int COUT = MT * 16;
  constexpr int CB   = CIN / 32;       // k-blocks per (dy,dx)
  constexpr int KB   = CIN * 9 / 32;   // total k-blocks
  constexpr int MGRP = MT / MW;        // wave groups along M
  constexpr int NGRP = 8 / MGRP;       // wave groups along N
  constexpr int NW   = 2;              // n-tiles per wave
  static_assert(MGRP * NGRP == 8, "8 waves per block");

  const int wave = threadIdx.x >> 5, lane = threadIdx.x & 31;
  const int mg = wave % MGRP, ng = wave / MGRP;
  const int n = lane & 15, hb = lane >> 4;

  int  p[NW], px[NW], py[NW];
  bool pv[NW];
  #pragma unroll
  for (int nw = 0; nw < NW; ++nw) {
    p[nw]  = blockIdx.x * (NGRP * NW * 16) + (ng * NW + nw) * 16 + n;
    pv[nw] = p[nw] < HWSZ;
    px[nw] = p[nw] % WW;
    py[nw] = p[nw] / WW;
  }

  v8f acc[MW][NW];
  #pragma unroll
  for (int mw = 0; mw < MW; ++mw)
    #pragma unroll
    for (int nw = 0; nw < NW; ++nw) acc[mw][nw] = (v8f){};

#if USE_ASYNC_LDS
  __shared__ _Float16 apanel[2][COUT * 32];
  constexpr int NTHR = (COUT * 32 * 2) / 16;   // b128 chunks per panel (64 or 256)
  const int tid = threadIdx.x;
  // issue one async DMA of panel `kb` into LDS buffer `buf`
  auto issue_panel = [&](int kb, int buf) {
    if (tid < NTHR) {
      const char* g = (const char*)(wpack + (size_t)kb * (COUT * 32)) + tid * 16;
      char* l = (char*)&apanel[buf][0] + tid * 16;
      __builtin_amdgcn_global_load_async_to_lds_b128((v4i*)g, (v4i*)l, 0, 0);
    }
  };
  issue_panel(0, 0);
#endif

  for (int kb = 0; kb < KB; ++kb) {
    const int r   = kb / CB;
    const int cib = (kb % CB) * 32;
    const int dxo = r % 3 - 1, dyo = r / 3 - 1;

#if USE_ASYNC_LDS
    const int buf = kb & 1;
    __builtin_amdgcn_s_wait_asynccnt(0);  // own panel chunks landed in LDS
    __syncthreads();                      // everyone's chunks landed; prev readers done
    if (kb + 1 < KB) issue_panel(kb + 1, buf ^ 1);
#endif

    v16h bfrag[NW];
    #pragma unroll
    for (int nw = 0; nw < NW; ++nw) {
      const int xx = px[nw] + dxo, yy = py[nw] + dyo;
      const bool v = pv[nw] && ((unsigned)xx < (unsigned)WW) && ((unsigned)yy < (unsigned)HH);
      bfrag[nw] = (v16h){};
      if (v) bfrag[nw] = *(const v16h*)(in_t + (size_t)(yy * WW + xx) * CIN + cib + 16 * hb);
    }

    v16h afrag[MW];
#if USE_ASYNC_LDS
    #pragma unroll
    for (int mw = 0; mw < MW; ++mw)
      afrag[mw] = *(const v16h*)&apanel[buf][((mg * MW + mw) * 32 + lane) * 16];
#else
    #pragma unroll
    for (int mw = 0; mw < MW; ++mw)
      afrag[mw] = *(const v16h*)(wpack + (((size_t)kb * MT + mg * MW + mw) * 32 + lane) * 16);
    if (kb + 1 < KB)  // gfx1250 global_prefetch_b8 on the streaming A operand
      __builtin_prefetch(wpack + (((size_t)(kb + 1) * MT + mg * MW) * 32 + lane) * 16, 0, 0);
#endif

    #pragma unroll
    for (int mw = 0; mw < MW; ++mw)
      #pragma unroll
      for (int nw = 0; nw < NW; ++nw)
        acc[mw][nw] = __builtin_amdgcn_wmma_f32_16x16x32_f16(
            false, afrag[mw], false, bfrag[nw], (short)0, acc[mw][nw], false, false);
  }

  // C/D layout: VGPR i -> row (i + 8*(lane>>4)), col = lane&15
  #pragma unroll
  for (int mw = 0; mw < MW; ++mw) {
    const int mt = mg * MW + mw;
    #pragma unroll
    for (int nw = 0; nw < NW; ++nw) {
      if constexpr (MODE == 0) {
        v8h o;
        #pragma unroll
        for (int i = 0; i < 8; ++i) {
          int c = mt * 16 + 8 * hb + i;
          float g = bnp[c], be = bnp[COUT + c], mu = bnp[2 * COUT + c], va = bnp[3 * COUT + c];
          float s = g * rsqrtf(va + 1e-5f);
          float val = acc[mw][nw][i] * s + (be - mu * s);
          o[i] = (_Float16)fmaxf(val, 0.f);
        }
        if (pv[nw]) *(v8h*)(out_t + (size_t)p[nw] * COUT + mt * 16 + 8 * hb) = o;  // 16B store
      } else {
        #pragma unroll
        for (int i = 0; i < 8; ++i) {
          int c = 8 * hb + i;
          if (c < NCLS && pv[nw]) {
            float val = acc[mw][nw][i] + bnp[c];
            if (do_sigmoid) val = 1.f / (1.f + expf(-val));
            out_f[(size_t)c * HWSZ + p[nw]] = val;
          }
        }
      }
    }
  }
}

// ---------------- top-k: 16-bit key histogram ----------------
__global__ void histo(const float* __restrict__ hm, unsigned* __restrict__ hist) {
  const int nn = NCLS * HWSZ;
  for (int i = blockIdx.x * blockDim.x + threadIdx.x; i < nn; i += gridDim.x * blockDim.x)
    atomicAdd(&hist[__float_as_uint(hm[i]) >> 16], 1u);   // positive floats: uint order == float order
}

// find threshold bin T: count(key > T) < TOPK <= count(key >= T)
__global__ __launch_bounds__(256)
void find_thresh(const unsigned* __restrict__ hist, unsigned* __restrict__ meta) {
  __shared__ unsigned csum[256];
  unsigned s = 0;
  for (int b = 0; b < 256; ++b) s += hist[threadIdx.x * 256 + b];
  csum[threadIdx.x] = s;
  __syncthreads();
  if (threadIdx.x == 0) {
    unsigned above = 0; int T = 0;
    for (int ct = 255; ct >= 0; --ct) {
      if (above + csum[ct] >= (unsigned)TOPK) {
        for (int b = ct * 256 + 255; b >= ct * 256; --b) {
          unsigned h = hist[b];
          if (above + h >= (unsigned)TOPK) { T = b; break; }
          above += h;
        }
        break;
      }
      above += csum[ct];
    }
    meta[0] = (unsigned)T;
    meta[1] = above;
  }
}

__global__ void compact(const float* __restrict__ hm, unsigned* __restrict__ meta,
                        unsigned* __restrict__ candA, unsigned* __restrict__ candB) {
  int i = blockIdx.x * blockDim.x + threadIdx.x;
  if (i >= NCLS * HWSZ) return;
  unsigned key = __float_as_uint(hm[i]) >> 16;
  unsigned T = meta[0];
  if (key > T) {
    unsigned pos = atomicAdd(&meta[2], 1u);
    if (pos < (unsigned)TOPK) candA[pos] = (unsigned)i;
  } else if (key == T) {
    unsigned pos = atomicAdd(&meta[3], 1u);
    if (pos < (unsigned)CAPB) candB[pos] = (unsigned)i;
  }
}

// exact rank selection (score desc, tie -> lower index) + fused decode
__global__ __launch_bounds__(512)
void select_decode(const float* __restrict__ outs, const unsigned* __restrict__ meta,
                   const unsigned* __restrict__ candA, const unsigned* __restrict__ candB,
                   float* __restrict__ dout) {
  __shared__ int   sidx[CAPB + TOPK];
  __shared__ float ssc [CAPB + TOPK];
  __shared__ int   selIdx[TOPK];
  __shared__ float selSc [TOPK];
  const float* hm = outs;  // branch 0 = sigmoid(hm), flat [3*HW] cls-major
  int nA = (int)meta[2]; if (nA > TOPK) nA = TOPK;
  int nB = (int)meta[3]; if (nB > CAPB) nB = CAPB;
  int M = nA + nB;
  for (int j = threadIdx.x; j < M; j += 512) {
    int gi = (j < nA) ? (int)candA[j] : (int)candB[j - nA];
    sidx[j] = gi; ssc[j] = hm[gi];
  }
  for (int t = threadIdx.x; t < TOPK; t += 512) { selIdx[t] = -1; selSc[t] = 0.f; }
  for (int t = threadIdx.x; t < 4500; t += 512) dout[t] = 0.f;
  __syncthreads();
  for (int j = threadIdx.x; j < M; j += 512) {
    float sj = ssc[j]; int ij = sidx[j];
    int rank = 0;
    for (int l = 0; l < M; ++l) {
      float sl = ssc[l];
      rank += (sl > sj) || (sl == sj && sidx[l] < ij);
    }
    if (rank < TOPK) { selIdx[rank] = ij; selSc[rank] = sj; }
  }
  __syncthreads();
  const float rect[3] = {0.68f, 0.71f, 0.65f};
  for (int r2 = threadIdx.x; r2 < TOPK; r2 += 512) {
    int idx = selIdx[r2];
    if (idx < 0) continue;
    float sc = selSc[r2];
    int cls = idx / HWSZ, sp = idx % HWSZ;
    float ys = (float)(sp / WW), xs = (float)(sp % WW);
    const float* o1 = outs + (size_t)1 * 3 * HWSZ;
    const float* o2 = outs + (size_t)2 * 3 * HWSZ;
    const float* o3 = outs + (size_t)3 * 3 * HWSZ;
    const float* o4 = outs + (size_t)4 * 3 * HWSZ;
    const float* o5 = outs + (size_t)5 * 3 * HWSZ;
    float c0 = o1[sp], c1 = o1[HWSZ + sp];
    float cz = o2[sp];
    float d0 = expf(o3[sp]), d1 = expf(o3[HWSZ + sp]), d2 = expf(o3[2 * HWSZ + sp]);
    float rc = o4[sp], rs = o4[HWSZ + sp];
    float iou = o5[sp];
    float xg = (xs + c0) * 0.4f - 75.2f;   // STRIDE*VOX = 4*0.1
    float yg = (ys + c1) * 0.4f - 75.2f;
    float heading = atan2f(rs, rc);
    iou = fminf(fmaxf((iou + 1.f) * 0.5f, 0.f), 1.f);
    float rr = rect[cls];
    float s2 = powf(sc, 1.f - rr) * powf(iou, rr);
    if (!(s2 > 0.1f)) s2 = 0.f;
    float* bx = dout + r2 * 7;
    bx[0] = xg; bx[1] = yg; bx[2] = cz; bx[3] = d0; bx[4] = d1; bx[5] = d2; bx[6] = heading;
    dout[3500 + r2] = s2;
    dout[4000 + r2] = (float)cls;
  }
}

// ---------------- host-side launch sequence ----------------
extern "C" void kernel_launch(void* const* d_in, const int* in_sizes, int n_in,
                              void* d_out, int out_size, void* d_ws, size_t ws_size,
                              hipStream_t stream) {
  const float* x    = (const float*)d_in[0];
  const float* Wsh  = (const float*)d_in[1];
  const float* bnsh = (const float*)d_in[2];
  const float* W1s  = (const float*)d_in[3];
  const float* bn1s = (const float*)d_in[4];
  const float* W2s  = (const float*)d_in[5];
  const float* b2s  = (const float*)d_in[6];
  float* dout = (float*)d_out;

  char* ws = (char*)d_ws;
  _Float16* xt    = (_Float16*)(ws + OFF_XT);
  _Float16* sht   = (_Float16*)(ws + OFF_SH);
  _Float16* brt   = (_Float16*)(ws + OFF_BR);
  float*    outs  = (float*)   (ws + OFF_OUTS);
  _Float16* wshp  = (_Float16*)(ws + OFF_WSH);
  _Float16* w1p   = (_Float16*)(ws + OFF_W1);
  _Float16* w2p   = (_Float16*)(ws + OFF_W2);
  unsigned* hist  = (unsigned*)(ws + OFF_HIST);
  unsigned* meta  = (unsigned*)(ws + OFF_META);
  unsigned* candA = (unsigned*)(ws + OFF_CA);
  unsigned* candB = (unsigned*)(ws + OFF_CB);

  // 0) zero histogram + meta counters (contiguous region)
  const int nz = 65536 + (int)(SZ_META / 4);
  zero_u32<<<(nz + 255) / 256, 256, 0, stream>>>(hist, nz);

  // 1) transpose + convert x
  dim3 tg((HWSZ + 31) / 32, CIN0 / 32);
  transpose_cvt<<<tg, 256, 0, stream>>>(x, xt);

  // 2) repack weights into A-fragment layout
  const int nWsh = CSH * CIN0 * 9;             // 147456
  pack_weights<<<(nWsh + 255) / 256, 256, 0, stream>>>(Wsh, wshp, CIN0, 4, CSH, nWsh, nWsh);
  const int pb1 = CSH * CSH * 9;               // 36864 / branch
  pack_weights<<<(6 * pb1 + 255) / 256, 256, 0, stream>>>(W1s, w1p, CSH, 4, CSH, pb1, 6 * pb1);
  const int pb2 = 16 * CSH * 9;                // 9216 / branch (M padded to 16)
  pack_weights<<<(6 * pb2 + 255) / 256, 256, 0, stream>>>(W2s, w2p, CSH, 1, NCLS, pb2, 6 * pb2);

  // 3) shared conv: 256 -> 64, BN + ReLU. Block tile 64x128 (2x2 reg blocking / wave)
  conv_wmma<CIN0, 4, 2, 0><<<(HWSZ + 127) / 128, 256, 0, stream>>>(
      xt, wshp, bnsh, sht, nullptr, 0);

  // 4) branches: (64->64 BN+ReLU) then (64->3 +bias, sigmoid for hm)
  for (int i = 0; i < 6; ++i) {
    conv_wmma<CSH, 4, 2, 0><<<(HWSZ + 127) / 128, 256, 0, stream>>>(
        sht, w1p + (size_t)i * pb1, bn1s + (size_t)i * 4 * CSH, brt, nullptr, 0);
    conv_wmma<CSH, 1, 1, 1><<<(HWSZ + 255) / 256, 256, 0, stream>>>(
        brt, w2p + (size_t)i * pb2, b2s + (size_t)i * 3, nullptr,
        outs + (size_t)i * 3 * HWSZ, (i == 0) ? 1 : 0);
  }

  // 5) exact top-500 + decode
  histo<<<1024, 256, 0, stream>>>(outs, hist);
  find_thresh<<<1, 256, 0, stream>>>(hist, meta);
  compact<<<(NCLS * HWSZ + 255) / 256, 256, 0, stream>>>(outs, meta, candA, candB);
  select_decode<<<1, 512, 0, stream>>>(outs, meta, candA, candB, dout);
}
